// PAM_Maps_Diff_8057358647907
// MI455X (gfx1250) — matrix operand order repaired and checked by reading, not compiled, not run on hardware
//
#include <hip/hip_runtime.h>

typedef __attribute__((ext_vector_type(16))) __bf16 v16bf;
typedef __attribute__((ext_vector_type(8)))  __bf16 v8bf;
typedef __attribute__((ext_vector_type(8)))  float  v8f;
typedef __attribute__((ext_vector_type(4)))  float  f32x4;

static constexpr int B_  = 8;
static constexpr int C_  = 512;
static constexpr int HW_ = 4096;    // 64*64

__device__ __forceinline__ v8f wmma_bf16(v16bf a, v16bf b, v8f c) {
    return __builtin_amdgcn_wmma_f32_16x16x32_bf16(
        /*neg_a=*/false, a, /*neg_b=*/false, b,
        /*c_mod=*/(short)0, c, /*reuse_a=*/false, /*reuse_b=*/false);
}

// ---------------------------------------------------------------------------
// K1: diff = map1-map2 ; feat_b = Wb*diff+bb ; feat_c = Wc*diff+bc
//     stored as [B*HW][6] (pixel-major) for cheap per-j gathers later.
// ---------------------------------------------------------------------------
__global__ void prep_bc_kernel(const float* __restrict__ map1,
                               const float* __restrict__ map2,
                               const float* __restrict__ Wb, const float* __restrict__ bb,
                               const float* __restrict__ Wc, const float* __restrict__ bc,
                               float* __restrict__ feat_b, float* __restrict__ feat_c)
{
    int p  = blockIdx.x * blockDim.x + threadIdx.x;   // 0 .. B*HW-1
    int b  = p >> 12;
    int hw = p & (HW_ - 1);
    float d[6];
#pragma unroll
    for (int c = 0; c < 6; ++c) {
        size_t idx = ((size_t)(b * 6 + c) << 12) + hw;
        d[c] = map1[idx] - map2[idx];
    }
#pragma unroll
    for (int o = 0; o < 6; ++o) {
        float vb = bb[o], vc = bc[o];
#pragma unroll
        for (int c = 0; c < 6; ++c) { vb += Wb[o * 6 + c] * d[c]; vc += Wc[o * 6 + c] * d[c]; }
        feat_b[(size_t)p * 6 + o] = vb;
        feat_c[(size_t)p * 6 + o] = vc;
    }
}

// ---------------------------------------------------------------------------
// K2: per attention row (b,i): rowmax = max_j S[i,j], rowinv = 1/sum_j exp(S-max)
//     One 256-thread block per row; 16 scores cached per thread in registers.
// ---------------------------------------------------------------------------
__global__ void softmax_stats_kernel(const float* __restrict__ feat_b,
                                     const float* __restrict__ feat_c,
                                     float* __restrict__ rowmax,
                                     float* __restrict__ rowinv)
{
    int row   = blockIdx.x;            // b*HW + i
    int tid   = threadIdx.x;
    int jbase = row & ~(HW_ - 1);      // b*HW
    const float* fb = feat_b + (size_t)row * 6;
    float f0 = fb[0], f1 = fb[1], f2 = fb[2], f3 = fb[3], f4 = fb[4], f5 = fb[5];

    float sc[16];
    float m = -3.402823466e38f;
#pragma unroll
    for (int t = 0; t < 16; ++t) {
        int j = tid + t * 256;
        const float* fc = feat_c + (size_t)(jbase + j) * 6;
        float s = f0 * fc[0] + f1 * fc[1] + f2 * fc[2] + f3 * fc[3] + f4 * fc[4] + f5 * fc[5];
        sc[t] = s;
        m = fmaxf(m, s);
    }

    __shared__ float red[256];
    red[tid] = m;
    __syncthreads();
    for (int off = 128; off > 0; off >>= 1) {
        if (tid < off) red[tid] = fmaxf(red[tid], red[tid + off]);
        __syncthreads();
    }
    m = red[0];
    __syncthreads();

    float sum = 0.f;
#pragma unroll
    for (int t = 0; t < 16; ++t) sum += __expf(sc[t] - m);
    red[tid] = sum;
    __syncthreads();
    for (int off = 128; off > 0; off >>= 1) {
        if (tid < off) red[tid] += red[tid + off];
        __syncthreads();
    }
    if (tid == 0) { rowmax[row] = m; rowinv[row] = 1.0f / red[0]; }
}

// ---------------------------------------------------------------------------
// K3: feat_d[b][c][j] = sum_k Wd[c][k]*feature[b][k][j] + bd[c], output bf16.
//     One 16(M=c) x 16(N=j) WMMA tile per wave, K=512 in steps of 32.
//     A built in contiguous k order (k = 16h + s), as is B; B gathered column-wise
//     (coalesced across the 16 lanes that share a k). Prefetch covers the
//     next K-step's strided feature columns.
// ---------------------------------------------------------------------------
__global__ void gemm_d_kernel(const float* __restrict__ feature,
                              const float* __restrict__ Wd,
                              const float* __restrict__ bd,
                              __bf16* __restrict__ featd)
{
    int wave = threadIdx.x >> 5;
    int L    = threadIdx.x & 31;
    int tile = blockIdx.x * 8 + wave;          // B*(C/16)*(HW/16) = 65536 tiles
    int b    = tile >> 13;                     // / (32*256)
    int r    = tile & 8191;
    int c0   = (r >> 8) << 4;                  // c tile origin
    int j0   = (r & 255) << 4;                 // j tile origin

    int arow  = c0 + (L & 15);                 // A-operand row (per-lane)
    int ksel  = (L < 16) ? 0 : 16;             // A-operand K sub-offset
    int bcol  = j0 + (L & 15);                 // B-operand column
    int bkoff = (L < 16) ? 0 : 16;             // B-operand K offset

    const float* wrow = Wd + (size_t)arow * C_;
    v8f acc = {};

    for (int k = 0; k < C_; k += 32) {
        // A: Wd[arow][k+ksel .. +15]  (f32 -> bf16)
        f32x4 a0 = *(const f32x4*)(wrow + k + ksel);
        f32x4 a1 = *(const f32x4*)(wrow + k + ksel + 4);
        f32x4 a2 = *(const f32x4*)(wrow + k + ksel + 8);
        f32x4 a3 = *(const f32x4*)(wrow + k + ksel + 12);
        v16bf A;
#pragma unroll
        for (int t = 0; t < 4; ++t) {
            A[t]      = (__bf16)a0[t];
            A[4 + t]  = (__bf16)a1[t];
            A[8 + t]  = (__bf16)a2[t];
            A[12 + t] = (__bf16)a3[t];
        }
        // B: feature[b][k+bkoff+t][bcol] for t=0..15 (column gather, lanes coalesce)
        const float* fp = feature + (size_t)(b * C_ + k + bkoff) * HW_ + bcol;
        v16bf Bv;
#pragma unroll
        for (int t = 0; t < 16; ++t) Bv[t] = (__bf16)fp[(size_t)t * HW_];

        // prefetch the next K-step's strided feature columns into cache
        if (k + 32 < C_) {
            const float* nfp = fp + (size_t)32 * HW_;
#pragma unroll
            for (int t = 0; t < 16; ++t) __builtin_prefetch(nfp + (size_t)t * HW_, 0, 3);
        }

        acc = wmma_bf16(A, Bv, acc);
    }

    int crow = c0 + ((L < 16) ? 0 : 8);
    int j    = j0 + (L & 15);
#pragma unroll
    for (int rr = 0; rr < 8; ++rr) {
        float v = acc[rr] + bd[crow + rr];
        featd[(size_t)(b * C_ + crow + rr) * HW_ + j] = (__bf16)v;
    }
}

// ---------------------------------------------------------------------------
// K4: fused softmax-apply + feat_e = feat_d @ attn^T + epilogue.
//     Workgroup: one (b, 16-row i-tile) and the full C=512 output strip.
//     8 waves x 4 accumulators (16x16 each) = 16 x 512 output tile.
//     Per 32-wide j step: P[16][32] = softmax tile built cooperatively in LDS
//     (K=6 score recompute + v_exp), then 4 WMMAs per wave with feat_d as B.
//     Next j-block's feat_d lines are prefetched behind the softmax VALU work.
// ---------------------------------------------------------------------------
__global__ void attn_out_kernel(const float* __restrict__ feat_b,
                                const float* __restrict__ feat_c,
                                const float* __restrict__ rowmax,
                                const float* __restrict__ rowinv,
                                const __bf16* __restrict__ featd,
                                const float* __restrict__ feature,
                                const float* __restrict__ alpha,
                                float* __restrict__ out)
{
    __shared__ __bf16 P[16][32];     // softmax tile (A operand source)
    __shared__ float  fbs[16][6];
    __shared__ float  rmaxs[16], rinvs[16];

    int tid  = threadIdx.x;
    int b    = blockIdx.x >> 8;
    int i0   = (blockIdx.x & 255) << 4;
    int rowb = b * HW_ + i0;

    if (tid < 96) fbs[tid / 6][tid % 6] = feat_b[(size_t)(rowb + tid / 6) * 6 + (tid % 6)];
    if (tid < 16) { rmaxs[tid] = rowmax[rowb + tid]; rinvs[tid] = rowinv[rowb + tid]; }
    __syncthreads();

    int wave  = tid >> 5;
    int L     = tid & 31;
    int il    = tid >> 4;            // P row this thread fills
    int jj    = (tid << 1) & 31;     // P cols (jj, jj+1)
    int prow  = L & 15;              // A-operand row per lane
    int abase = (L < 16) ? 0 : 16;   // A-operand K sub-offset
    int bkoff = (L < 16) ? 0 : 16;   // B-operand K offset
    size_t bbase = (size_t)b * C_ * HW_;
    const __bf16* bptr0 = featd + bbase + (size_t)(wave * 64 + (L & 15)) * HW_ + bkoff;
    int jcbase = b * HW_;

    // hoist loop invariants out of the 128-iteration hot loop (kills 8 ds
    // loads per thread per iteration)
    float fb0 = fbs[il][0], fb1 = fbs[il][1], fb2 = fbs[il][2];
    float fb3 = fbs[il][3], fb4 = fbs[il][4], fb5 = fbs[il][5];
    float rm  = rmaxs[il],  ri  = rinvs[il];

    v8f acc0 = {}, acc1 = {}, acc2 = {}, acc3 = {};

    for (int jb = 0; jb < HW_; jb += 32) {
        // --- build P tile: 2 softmax values per thread ---
        const float* fc = feat_c + (size_t)(jcbase + jb + jj) * 6;
        float s0 = fb0 * fc[0] + fb1 * fc[1] + fb2 * fc[2]
                 + fb3 * fc[3] + fb4 * fc[4] + fb5 * fc[5];
        float s1 = fb0 * fc[6] + fb1 * fc[7] + fb2 * fc[8]
                 + fb3 * fc[9] + fb4 * fc[10] + fb5 * fc[11];
        P[il][jj]     = (__bf16)(__expf(s0 - rm) * ri);
        P[il][jj + 1] = (__bf16)(__expf(s1 - rm) * ri);

        // prefetch next j-block's feat_d lines while softmax settles
        if (jb + 32 < HW_) {
            const __bf16* np = bptr0 + jb + 32;
            __builtin_prefetch(np, 0, 3);
            __builtin_prefetch(np + (size_t)16 * HW_, 0, 3);
            __builtin_prefetch(np + (size_t)32 * HW_, 0, 3);
            __builtin_prefetch(np + (size_t)48 * HW_, 0, 3);
        }
        __syncthreads();

        // --- A operand from LDS, contiguous k order (k = 16h + s) ---
        v8bf lo = *(const v8bf*)&P[prow][abase];
        v8bf hi = *(const v8bf*)&P[prow][abase + 8];
        v16bf A = __builtin_shufflevector(lo, hi, 0, 1, 2, 3, 4, 5, 6, 7,
                                          8, 9, 10, 11, 12, 13, 14, 15);
        __syncthreads();   // P consumed; safe to overwrite next iteration

        // --- 4 x WMMA: B operand = contiguous 32B of feat_d per lane ---
        const __bf16* bp = bptr0 + jb;
        acc0 = wmma_bf16(A, *(const v16bf*)bp, acc0);
        acc1 = wmma_bf16(A, *(const v16bf*)(bp + (size_t)16 * HW_), acc1);
        acc2 = wmma_bf16(A, *(const v16bf*)(bp + (size_t)32 * HW_), acc2);
        acc3 = wmma_bf16(A, *(const v16bf*)(bp + (size_t)48 * HW_), acc3);
    }

    // --- epilogue: out = alpha*feat_e + feature_map ---
    // per lane the 8 accumulator rows map to 8 consecutive i's -> compiler can
    // merge into b128 load/store pairs
    float al   = alpha[0];
    int   ioff = i0 + ((L < 16) ? 0 : 8);
    auto store_tile = [&](const v8f& a, int t) {
        size_t cidx = bbase + (size_t)(wave * 64 + t * 16 + (L & 15)) * HW_;
#pragma unroll
        for (int rr = 0; rr < 8; ++rr) {
            size_t idx = cidx + ioff + rr;
            out[idx] = al * a[rr] + feature[idx];
        }
    };
    store_tile(acc0, 0);
    store_tile(acc1, 1);
    store_tile(acc2, 2);
    store_tile(acc3, 3);
}

// ---------------------------------------------------------------------------
extern "C" void kernel_launch(void* const* d_in, const int* in_sizes, int n_in,
                              void* d_out, int out_size, void* d_ws, size_t ws_size,
                              hipStream_t stream)
{
    (void)in_sizes; (void)n_in; (void)out_size; (void)ws_size;

    const float* map1    = (const float*)d_in[0];
    const float* map2    = (const float*)d_in[1];
    const float* feature = (const float*)d_in[2];
    const float* Wb      = (const float*)d_in[3];
    const float* bb      = (const float*)d_in[4];
    const float* Wc      = (const float*)d_in[5];
    const float* bc      = (const float*)d_in[6];
    const float* Wd      = (const float*)d_in[7];
    const float* bd      = (const float*)d_in[8];
    const float* alpha   = (const float*)d_in[9];
    float*       out     = (float*)d_out;

    // workspace carve (all sections 256B aligned)
    char*   w      = (char*)d_ws;
    float*  feat_b = (float*)w;  w += (size_t)B_ * HW_ * 6 * sizeof(float);   // 768 KB
    float*  feat_c = (float*)w;  w += (size_t)B_ * HW_ * 6 * sizeof(float);   // 768 KB
    float*  rmax   = (float*)w;  w += (size_t)B_ * HW_ * sizeof(float);       // 128 KB
    float*  rinv   = (float*)w;  w += (size_t)B_ * HW_ * sizeof(float);       // 128 KB
    __bf16* featd  = (__bf16*)w;                                              // 32 MB

    prep_bc_kernel<<<(B_ * HW_) / 256, 256, 0, stream>>>(map1, map2, Wb, bb, Wc, bc,
                                                         feat_b, feat_c);
    softmax_stats_kernel<<<B_ * HW_, 256, 0, stream>>>(feat_b, feat_c, rmax, rinv);
    gemm_d_kernel<<<(B_ * (C_ / 16) * (HW_ / 16)) / 8, 256, 0, stream>>>(feature, Wd, bd, featd);
    attn_out_kernel<<<B_ * (HW_ / 16), 256, 0, stream>>>(feat_b, feat_c, rmax, rinv,
                                                         featd, feature, alpha, out);
}
